// MultiheadAttention_81724637708415
// MI455X (gfx1250) — matrix operand order repaired and checked by reading, not compiled, not run on hardware
//
#include <hip/hip_runtime.h>
#include <hip/hip_bf16.h>

typedef __attribute__((ext_vector_type(16))) _Float16 v16h;
typedef __attribute__((ext_vector_type(8)))  _Float16 v8h;
typedef __attribute__((ext_vector_type(4)))  _Float16 v4h;
typedef __attribute__((ext_vector_type(8)))  float    v8f;

#define EMBED   1024
#define HEADS   16
#define HDIM    64
#define BATCHSZ 4
#define SEQ     2048
#define KDIM    EMBED

#ifndef __has_builtin
#define __has_builtin(x) 0
#endif
#if __has_builtin(__builtin_amdgcn_tensor_load_to_lds) && \
    __has_builtin(__builtin_amdgcn_s_wait_tensorcnt)
#define USE_TDM 1
#else
#define USE_TDM 0
#endif

// Raw v_exp_f32 / v_rcp_f32: args are <= 0 (softmax-shifted), denormal flush
// to zero is exactly the desired behavior; avoids the libm range-fix expansion.
#if __has_builtin(__builtin_amdgcn_exp2f)
#define EXP2F(x) __builtin_amdgcn_exp2f(x)
#else
#define EXP2F(x) exp2f(x)
#endif
#if __has_builtin(__builtin_amdgcn_rcpf)
#define RCPF(x) __builtin_amdgcn_rcpf(x)
#else
#define RCPF(x) (1.0f / (x))
#endif

union FragAB {
  v16h v;
  v8h  h[2];
};

__device__ __forceinline__ v8f wmma_f16(const FragAB& a, const FragAB& b, v8f c) {
  return __builtin_amdgcn_wmma_f32_16x16x32_f16(
      false, a.v, false, b.v, (short)0, c, false, false);
}

__device__ __forceinline__ v8f v8f_zero() {
  v8f z = {0.f, 0.f, 0.f, 0.f, 0.f, 0.f, 0.f, 0.f};
  return z;
}

#if USE_TDM
typedef __attribute__((ext_vector_type(4))) unsigned tdm_v4u;
typedef __attribute__((ext_vector_type(8))) int      tdm_v8i;
typedef __attribute__((ext_vector_type(4))) int      tdm_v4i;

// TDM: DMA a [rows x cols] f16 tile (row stride = strideElems) from global
// into LDS at byte offset ldsOff, inserting 4 DWORDs of pad every 32 DWORDs
// (i.e. 8 pad halves per 64-half row -> 72-half LDS row stride).
__device__ __forceinline__ void tdm_load_tile_f16(const _Float16* gsrc,
                                                  unsigned ldsOff,
                                                  int rows, int cols,
                                                  int strideElems)
{
  unsigned long long ga = (unsigned long long)(uintptr_t)gsrc;
  tdm_v4u g0;
  g0[0] = 1u;                                            // count=1, no gather
  g0[1] = ldsOff;                                        // lds_addr (bytes)
  g0[2] = (unsigned)(ga & 0xFFFFFFFFull);                // global_addr lo
  g0[3] = (unsigned)((ga >> 32) & 0x01FFFFFFull)         // global_addr hi
          | 0x80000000u;                                 // type=2 (image)
  tdm_v8i g1;
  g1[0] = (int)(0x00010000u        // data_size=1 (2B)
          | (1u << 20)             // pad_enable
          | (4u << 22)             // pad_interval: 32 DWORDs
          | (3u << 25));           // pad_amount: 4 DWORDs (8 halves)
  g1[1] = (int)(((unsigned)cols & 0xFFFFu) << 16);       // tensor_dim0 lo16
  g1[2] = (int)((((unsigned)cols >> 16) & 0xFFFFu)       // tensor_dim0 hi16
          | (((unsigned)SEQ & 0xFFFFu) << 16));          // tensor_dim1 lo16
  g1[3] = (int)((((unsigned)SEQ >> 16) & 0xFFFFu)        // tensor_dim1 hi16
          | (((unsigned)cols & 0xFFFFu) << 16));         // tile_dim0
  g1[4] = (int)((unsigned)rows & 0xFFFFu);               // tile_dim1 (dim2=0)
  g1[5] = strideElems;                                   // dim0_stride lo32
  g1[6] = 0;                                             // stride hi / dim1_stride
  g1[7] = 0;
  tdm_v4i gz = {0, 0, 0, 0};
#if __clang_major__ >= 23
  tdm_v8i gz8 = {0, 0, 0, 0, 0, 0, 0, 0};
  __builtin_amdgcn_tensor_load_to_lds(g0, g1, gz, gz, gz8, 0);
#else
  __builtin_amdgcn_tensor_load_to_lds(g0, g1, gz, gz, 0);
#endif
}
#endif

// ---------------------------------------------------------------------------
// Y = (A * W^T + b) * oscale.  A: [M,K] f32 or f16, W: [N,K] f32.
// 128x128 block tile, 8 waves, each wave 64x32 (4x2 WMMA accumulators).
// MODE 0: f16 out with [B,H,S,Dh] head-split permutation.
// MODE 1: f32 out row-major [M,N].
// ---------------------------------------------------------------------------
template<bool A_HALF, int MODE>
__global__ __launch_bounds__(256, 2)
void gemm_xwt(const void* __restrict__ Aptr,
              const float* __restrict__ W,
              const float* __restrict__ bias,
              _Float16* __restrict__ outH,
              float* __restrict__ outF,
              int M, int N, int K, float oscale)
{
  __shared__ alignas(16) _Float16 As[128][40];   // 32 + 8 pad halves
  __shared__ alignas(16) _Float16 Ws[128][40];

  const int tid  = threadIdx.x;
  const int lane = tid & 31;
  const int wid  = tid >> 5;
  const int wr   = wid >> 2;
  const int wc   = wid & 3;
  const int half = lane >> 4;
  const int ln   = lane & 15;

  const int m0 = blockIdx.y * 128;
  const int n0 = blockIdx.x * 128;

  v8f acc[4][2];
  #pragma unroll
  for (int i = 0; i < 4; ++i)
    #pragma unroll
    for (int j = 0; j < 2; ++j)
      acc[i][j] = v8f_zero();

  const float*    Af = (const float*)Aptr;
  const _Float16* Ah = (const _Float16*)Aptr;

  for (int kk = 0; kk < K; kk += 32) {
    if (A_HALF) {
      #pragma unroll
      for (int t = 0; t < 2; ++t) {
        int idx = tid + t * 256;
        int r = idx >> 2;
        int c = (idx & 3) * 8;
        *(v8h*)&As[r][c] = *(const v8h*)(Ah + (size_t)(m0 + r) * K + kk + c);
      }
    } else {
      #pragma unroll
      for (int t = 0; t < 4; ++t) {
        int idx = tid + t * 256;
        int r = idx >> 3;
        int c = (idx & 7) * 4;
        float4 f = *(const float4*)(Af + (size_t)(m0 + r) * K + kk + c);
        v4h hv;
        hv[0] = (_Float16)f.x; hv[1] = (_Float16)f.y;
        hv[2] = (_Float16)f.z; hv[3] = (_Float16)f.w;
        *(v4h*)&As[r][c] = hv;
      }
    }
    #pragma unroll
    for (int t = 0; t < 4; ++t) {
      int idx = tid + t * 256;
      int r = idx >> 3;
      int c = (idx & 7) * 4;
      float4 f = *(const float4*)(W + (size_t)(n0 + r) * K + kk + c);
      v4h hv;
      hv[0] = (_Float16)f.x; hv[1] = (_Float16)f.y;
      hv[2] = (_Float16)f.z; hv[3] = (_Float16)f.w;
      *(v4h*)&Ws[r][c] = hv;
    }
    if (kk + 32 < K) {
      __builtin_prefetch(W + (size_t)(n0 + (tid >> 1)) * K + kk + 32, 0, 0);
    }
    __syncthreads();

    FragAB a[4], b[2];
    #pragma unroll
    for (int mf = 0; mf < 4; ++mf) {
      const _Float16* arow = &As[wr * 64 + mf * 16 + ln][0];
      a[mf].h[0] = *(const v8h*)(arow + 8 * half);
      a[mf].h[1] = *(const v8h*)(arow + 16 + 8 * half);
    }
    #pragma unroll
    for (int nf = 0; nf < 2; ++nf) {
      const _Float16* wrow = &Ws[wc * 32 + nf * 16 + ln][0];
      b[nf].h[0] = *(const v8h*)(wrow + 8 * half);
      b[nf].h[1] = *(const v8h*)(wrow + 16 + 8 * half);
    }
    #pragma unroll
    for (int mf = 0; mf < 4; ++mf)
      #pragma unroll
      for (int nf = 0; nf < 2; ++nf)
        acc[mf][nf] = wmma_f16(a[mf], b[nf], acc[mf][nf]);
    __syncthreads();
  }

  #pragma unroll
  for (int mf = 0; mf < 4; ++mf) {
    #pragma unroll
    for (int nf = 0; nf < 2; ++nf) {
      const int col = n0 + wc * 32 + nf * 16 + ln;
      const float bv = bias[col];
      #pragma unroll
      for (int r = 0; r < 8; ++r) {
        const int row = m0 + wr * 64 + mf * 16 + r + 8 * half;
        if (MODE == 0) {
          const float v = (acc[mf][nf][r] + bv) * oscale;
          const int hh = col >> 6, d = col & (HDIM - 1);
          const int bb = row >> 11, s = row & (SEQ - 1);
          outH[(((size_t)(bb * HEADS + hh) * SEQ) + s) * HDIM + d] = (_Float16)v;
        } else {
          outF[(size_t)row * N + col] = acc[mf][nf][r] + bv;
        }
      }
    }
  }
}

// ---------------------------------------------------------------------------
// Flash attention. Q rows pre-scaled by (1/sqrt(Dh))*log2(e) so softmax runs
// entirely in the base-2 domain (bare v_exp_f32). Row sums computed by the
// tensor core via an all-ones block appended to V^T (5th accumulator).
// K tiles staged by the Tensor Data Mover when available.
// ---------------------------------------------------------------------------
__global__ __launch_bounds__(128, 2)
void flash_attn(const _Float16* __restrict__ Q,
                const _Float16* __restrict__ Kg,
                const _Float16* __restrict__ Vg,
                _Float16* __restrict__ O)
{
  __shared__ alignas(16) _Float16 Ks[64][72];        // [kv][d]
  __shared__ alignas(16) _Float16 Vt[80][72];        // [d][kv]; rows 64..79 = ones
  __shared__ alignas(16) _Float16 Ps[4][16][72];     // per-wave P staging

  const int tid  = threadIdx.x;
  const int lane = tid & 31;
  const int wid  = tid >> 5;
  const int half = lane >> 4;
  const int ln   = lane & 15;

  const int bh = blockIdx.y;
  const int q0 = blockIdx.x * 64;
  const int bb = bh >> 4;
  const int hh = bh & 15;

  const size_t base = (size_t)bh * SEQ * HDIM;

  // ones block for tensor-core row sums (written once, never overwritten)
  #pragma unroll
  for (int t = 0; t < 8; ++t) {
    int idx = tid + t * 128;
    Vt[64 + (idx >> 6)][idx & 63] = (_Float16)1.0f;
  }

  FragAB qf[2];
  {
    const _Float16* qrow = Q + base + (size_t)(q0 + wid * 16 + ln) * HDIM;
    #pragma unroll
    for (int ks = 0; ks < 2; ++ks) {
      qf[ks].h[0] = *(const v8h*)(qrow + ks * 32 + 8 * half);
      qf[ks].h[1] = *(const v8h*)(qrow + ks * 32 + 16 + 8 * half);
    }
  }

  v8f o[5];                      // o[0..3] = O tile, o[4] = row sums l
  #pragma unroll
  for (int nf = 0; nf < 5; ++nf) o[nf] = v8f_zero();
  float mrow[8];
  #pragma unroll
  for (int r = 0; r < 8; ++r) mrow[r] = -3.0e38f;

#if USE_TDM
  typedef __attribute__((address_space(3))) _Float16 lds_f16;
  const unsigned ksOff = (unsigned)(uintptr_t)(lds_f16*)&Ks[0][0];
#endif

  for (int kv0 = 0; kv0 < SEQ; kv0 += 64) {
#if USE_TDM
    if (wid == 0)
      tdm_load_tile_f16(Kg + base + (size_t)kv0 * HDIM, ksOff, 64, HDIM, HDIM);
#else
    #pragma unroll
    for (int t = 0; t < 4; ++t) {
      int idx = tid + t * 128;
      int r = idx >> 3;
      int c = (idx & 7) * 8;
      *(v8h*)&Ks[r][c] = *(const v8h*)(Kg + base + (size_t)(kv0 + r) * HDIM + c);
    }
#endif
    // V tile, hand-transposed into Vt[d][kv]
    #pragma unroll
    for (int t = 0; t < 4; ++t) {
      int idx = tid + t * 128;
      int r = idx >> 3;
      int c = (idx & 7) * 8;
      v8h vv = *(const v8h*)(Vg + base + (size_t)(kv0 + r) * HDIM + c);
      #pragma unroll
      for (int j = 0; j < 8; ++j) Vt[c + j][r] = vv[j];
    }
#if USE_TDM
    if (wid == 0) __builtin_amdgcn_s_wait_tensorcnt(0);
#endif
    __syncthreads();

    // ---- S = Q' K^T  (scale already folded into Q) ----
    v8f s[4];
    #pragma unroll
    for (int nf = 0; nf < 4; ++nf) s[nf] = v8f_zero();
    #pragma unroll
    for (int ks = 0; ks < 2; ++ks) {
      #pragma unroll
      for (int nf = 0; nf < 4; ++nf) {
        FragAB kb;
        const _Float16* krow = &Ks[nf * 16 + ln][ks * 32 + 8 * half];
        kb.h[0] = *(const v8h*)krow;
        kb.h[1] = *(const v8h*)(krow + 16);
        s[nf] = wmma_f16(qf[ks], kb, s[nf]);
      }
    }

    // ---- online softmax, base-2 domain; only the max needs shuffles ----
    float alpha[8];
    #pragma unroll
    for (int r = 0; r < 8; ++r) {
      float mx = fmaxf(fmaxf(s[0][r], s[1][r]), fmaxf(s[2][r], s[3][r]));
      #pragma unroll
      for (int off = 8; off >= 1; off >>= 1)
        mx = fmaxf(mx, __shfl_xor(mx, off, 16));
      const float mn = fmaxf(mrow[r], mx);
      alpha[r] = EXP2F(mrow[r] - mn);
      mrow[r] = mn;
      #pragma unroll
      for (int nf = 0; nf < 4; ++nf)
        s[nf][r] = EXP2F(s[nf][r] - mn);
      #pragma unroll
      for (int nf = 0; nf < 5; ++nf)      // rescale O and the row-sum acc
        o[nf][r] *= alpha[r];
    }

    // ---- restage P: C-layout -> LDS -> A-layout ----
    #pragma unroll
    for (int nf = 0; nf < 4; ++nf)
      #pragma unroll
      for (int r = 0; r < 8; ++r)
        Ps[wid][r + 8 * half][nf * 16 + ln] = (_Float16)s[nf][r];
    __syncthreads();

    // ---- O += P * [V | 1]  (nf==4 accumulates row sums) ----
    #pragma unroll
    for (int ks2 = 0; ks2 < 2; ++ks2) {
      FragAB pa;
      const _Float16* prow = &Ps[wid][ln][ks2 * 32];
      pa.h[0] = *(const v8h*)(prow + 8 * half);
      pa.h[1] = *(const v8h*)(prow + 16 + 8 * half);
      #pragma unroll
      for (int nf = 0; nf < 5; ++nf) {
        FragAB vb;
        const _Float16* vrow = &Vt[nf * 16 + ln][ks2 * 32 + 8 * half];
        vb.h[0] = *(const v8h*)vrow;
        vb.h[1] = *(const v8h*)(vrow + 16);
        o[nf] = wmma_f16(pa, vb, o[nf]);
      }
    }
    __syncthreads();
  }

  // ---- normalize by tensor-core row sums; write f16 [B,S,D] ----
  #pragma unroll
  for (int r = 0; r < 8; ++r) {
    const float inv = RCPF(o[4][r]);
    const int row = q0 + wid * 16 + r + 8 * half;
    #pragma unroll
    for (int nf = 0; nf < 4; ++nf) {
      const int col = hh * HDIM + nf * 16 + ln;
      O[((size_t)bb * SEQ + row) * EMBED + col] = (_Float16)(o[nf][r] * inv);
    }
  }
}

// ---------------------------------------------------------------------------
extern "C" void kernel_launch(void* const* d_in, const int* in_sizes, int n_in,
                              void* d_out, int out_size, void* d_ws, size_t ws_size,
                              hipStream_t stream)
{
  const float* query = (const float*)d_in[0];
  const float* key   = (const float*)d_in[1];
  const float* value = (const float*)d_in[2];
  const float* Wq    = (const float*)d_in[3];
  const float* bq    = (const float*)d_in[4];
  const float* Wk    = (const float*)d_in[5];
  const float* bk    = (const float*)d_in[6];
  const float* Wv    = (const float*)d_in[7];
  const float* bv    = (const float*)d_in[8];
  const float* Wo    = (const float*)d_in[9];
  const float* bo    = (const float*)d_in[10];

  const int M = BATCHSZ * SEQ;
  const size_t elems = (size_t)M * EMBED;

  _Float16* qh = (_Float16*)d_ws;
  _Float16* kh = qh + elems;
  _Float16* vh = kh + elems;
  _Float16* oh = vh + elems;

  dim3 ggrid(EMBED / 128, M / 128);
  dim3 gblk(256);

  // Fold softmax scale and log2(e) into the Q projection epilogue:
  // scores in base-2 domain -> bare v_exp_f32 in the flash kernel.
  const float qscale = 0.125f * 1.44269504088896f;   // 1/sqrt(64) * log2(e)

  gemm_xwt<false, 0><<<ggrid, gblk, 0, stream>>>(query, Wq, bq, qh, nullptr, M, EMBED, KDIM, qscale);
  gemm_xwt<false, 0><<<ggrid, gblk, 0, stream>>>(key,   Wk, bk, kh, nullptr, M, EMBED, KDIM, 1.0f);
  gemm_xwt<false, 0><<<ggrid, gblk, 0, stream>>>(value, Wv, bv, vh, nullptr, M, EMBED, KDIM, 1.0f);

  flash_attn<<<dim3(SEQ / 64, BATCHSZ * HEADS), 128, 0, stream>>>(qh, kh, vh, oh);

  gemm_xwt<true, 1><<<ggrid, gblk, 0, stream>>>(oh, Wo, bo, nullptr, (float*)d_out, M, EMBED, KDIM, 1.0f);
}
